// ChirpLinker_30623116820528
// MI455X (gfx1250) — compile-verified, matrix operand order checked by reading, not run on hardware
//
#include <hip/hip_runtime.h>
#include <math.h>

// ChirpLinker for MI455X (gfx1250, wave32).
// One workgroup per batch; membership as bitmask; selection matvecs via
// v_wmma_f32_16x16x32_f16 (A rows 0/1 = rank+1 / selected-total vectors,
// B = on-the-fly unpacked membership bits).

typedef __attribute__((ext_vector_type(16))) _Float16 v16h;
typedef __attribute__((ext_vector_type(8)))  _Float16 v8h;
typedef __attribute__((ext_vector_type(8)))  float    v8f;
typedef unsigned int u32;

#define BB   4
#define WW   64
#define KK   64
#define NTOK 4096           // WW*KK
#define NWRD 128            // bitmask words per membership row
#define THREADS 256

union V16U { v16h v; _Float16 e[16]; v8h h8[2]; u32 w[8]; };

__device__ __forceinline__ float wrapf(float x) {
    const float PIF = 3.14159265358979323846f;
    const float TPF = 6.28318530717958647692f;
    float y = fmodf(x + PIF, TPF);
    y = (y < 0.f) ? y + TPF : y;
    return y - PIF;
}

__global__ __launch_bounds__(THREADS)
void chirp_link_batch(const float* __restrict__ tok,   // (B,N,9)
                      float* __restrict__ outData,     // (B,N,9)
                      u32*   __restrict__ memberW,     // B*N*128 words
                      float* __restrict__ bestW,       // B*N
                      int*   __restrict__ predW,       // B*N
                      float* __restrict__ totalW,      // B*N
                      int*   __restrict__ lenW,        // B*N
                      float* __restrict__ idp1W,       // B*N
                      float* __restrict__ ttlW,        // B*N
                      int*   __restrict__ cidW,        // B*N
                      int*   __restrict__ nqW)         // B
{
    __shared__ float    sKey[NTOK];     // sort keys (total), becomes sorted
    __shared__ int      sIdx[NTOK];     // sort payload, becomes order[]
    __shared__ __align__(32) _Float16 sRank[NTOK];  // ordered row: rank+1 (or 0)
    __shared__ __align__(32) _Float16 sTsel[NTOK];  // ordered row: qual ? total : 0
    __shared__ __align__(32) _Float16 sZero[32];    // zero pad for A rows >= 2
    __shared__ float    sPB[KK], sPFe[KK], sPAe[KK], sPPe[KK]; // prev window

    const int b    = blockIdx.x;
    const int tid  = threadIdx.x;
    const int lane = tid & 31;
    const int wv   = tid >> 5;
    const float* T = tok + (size_t)b * NTOK * 9;
    const size_t bo = (size_t)b * NTOK;

    if (tid < 32) sZero[tid] = (_Float16)0.f;

    // ---------------- Phase A: DP forward scan ----------------
    if (tid < KK) {
        int n = tid;
        float s  = T[n*9+0];
        float b0 = (s > 0.f) ? s*s : -__builtin_inff();
        bestW[bo+n] = b0;  predW[bo+n] = -1;
        sPB[tid] = b0; sPFe[tid] = T[n*9+4]; sPAe[tid] = T[n*9+6]; sPPe[tid] = T[n*9+8];
    }
    for (int w = 1; w < WW; ++w) {
        __syncthreads();
        float nb = 0.f, nfe = 0.f, nae = 0.f, npe = 0.f;
        bool act = (tid < KK);
        if (act) {
            if (w + 1 < WW)  // warm L2/WGP$ for next window (global_prefetch_b8)
                __builtin_prefetch(&T[((w + 1) * KK + tid) * 9], 0, 3);
            int n = w*KK + tid;
            float s  = T[n*9+0];
            float fs = T[n*9+3], As = T[n*9+5], ps = T[n*9+7];
            bool  posn = (s > 0.f);
            float bestv = -__builtin_inff(); int bp = -1;
            for (int j = 0; j < KK; ++j) {
                float pb  = sPB[j];
                float fep = sPFe[j];
                float fm  = 0.5f * (fep + fs);
                bool bad_f = (fm > 0.f) && (fabsf(fep - fs) / ((fm > 0.f) ? fm : 1.f) > 0.05f);
                float dphi = wrapf(ps - sPPe[j]);
                bool bad_p = fabsf(dphi) > 0.5f;
                float ae  = sPAe[j];
                float am  = fmaxf(ae, As);
                bool bad_a = (am > 0.f) && (fabsf(ae - As) / ((am > 0.f) ? am : 1.f) > 0.5f);
                bool comp = posn && !bad_f && !bad_p && !bad_a;
                float val = comp ? pb : -__builtin_inff();
                if (val > bestv) { bestv = val; bp = j; }
            }
            bool defined = (bestv != -__builtin_inff());
            float bn = defined ? (bestv + s*s) : -__builtin_inff();
            bestW[bo+n] = bn;
            predW[bo+n] = defined ? ((w-1)*KK + bp) : -1;
            nb = bn; nfe = T[n*9+4]; nae = T[n*9+6]; npe = T[n*9+8];
        }
        __syncthreads();
        if (act) { sPB[tid] = nb; sPFe[tid] = nfe; sPAe[tid] = nae; sPPe[tid] = npe; }
    }
    __syncthreads();

    // ------- Phase B: totals, lengths, sort keys, membership bitmask -------
    for (int n = tid; n < NTOK; n += THREADS) {
        float s  = T[n*9+0];
        float bv = bestW[bo+n];
        bool defined = (bv != -__builtin_inff());
        bool exists  = (s > 0.f);
        float tt = exists ? (defined ? bv : s*s) : -__builtin_inff();
        totalW[bo+n] = tt;
        lenW[bo+n]   = defined ? ((n >> 6) + 1) : 1;
        sKey[n] = tt;  sIdx[n] = n;

        u32* row = memberW + (bo + n) * NWRD;
        for (int wd = 0; wd < NWRD; ++wd) row[wd] = 0u;
        int cur = n;
        for (int st = 0; st < WW && cur >= 0; ++st) {
            row[cur >> 5] |= 1u << (cur & 31);
            cur = predW[bo + cur];
        }
    }
    __threadfence_block();
    __syncthreads();

    // ------- Phase C: bitonic sort by (-total, idx) -------
    for (int k = 2; k <= NTOK; k <<= 1) {
        for (int j = k >> 1; j > 0; j >>= 1) {
            __syncthreads();
            for (int i = tid; i < NTOK; i += THREADS) {
                int ixj = i ^ j;
                if (ixj > i) {
                    float ka = sKey[i],  kb2 = sKey[ixj];
                    int   ia = sIdx[i],  ib  = sIdx[ixj];
                    bool aFirst = (ka > kb2) || (ka == kb2 && ia < ib);
                    bool up = ((i & k) == 0);
                    if (up ? !aFirst : aFirst) {
                        sKey[i] = kb2; sKey[ixj] = ka;
                        sIdx[i] = ib;  sIdx[ixj] = ia;
                    }
                }
            }
        }
    }
    __syncthreads();

    // ------- Phase D: greedy non-overlap selection (wave 0 only) -------
    if (tid < 32) {
        u32 u0 = 0, u1 = 0, u2 = 0, u3 = 0;   // 4096-bit used mask in regs
        int cnt = 0;
        for (int o = 0; o < NTOK; ++o) {
            int   n  = sIdx[o];
            float tt = sKey[o];
            bool  ex = (tt != -__builtin_inff());
            const u32* row = memberW + (bo + n) * NWRD;
            u32 r0 = row[lane], r1 = row[lane+32], r2 = row[lane+64], r3 = row[lane+96];
            u32 ov = (r0 & u0) | (r1 & u1) | (r2 & u2) | (r3 & u3);
            for (int off = 16; off > 0; off >>= 1) ov |= __shfl_xor(ov, off, 32);
            bool sel = ex && (ov == 0u);
            if (sel) { u0 |= r0; u1 |= r1; u2 |= r2; u3 |= r3; }
            bool qual = sel && (lenW[bo + n] >= 2);
            if (qual) cnt++;
            if (lane == 0) {
                sRank[o] = (_Float16)(qual ? (float)cnt : 0.f);
                sTsel[o] = (_Float16)(qual ? tt : 0.f);
            }
        }
        if (lane == 0) nqW[b] = cnt;
    }
    __syncthreads();

    // ------- Phase E: fused matvecs via WMMA f32_16x16x32_f16 -------
    // idp1[n] = sum_o rank1[o]*member_o[o][n];  ttl[n] = sum_o tsel[o]*member_o[o][n]
    {
        const int m  = lane & 15;               // A row M (only 0/1 carry data)
        const int kb = (lane < 16) ? 0 : 8;     // K group base per ISA layout
        // Branchless A source: rows 0/1 read real vectors, others a zero pad.
        const _Float16* abase = (m == 0) ? sRank : ((m == 1) ? sTsel : sZero);
        const bool      areal = (m < 2);
        const int       col   = lane & 15;      // B column N
        for (int t = wv; t < NTOK / 16; t += (THREADS / 32)) {
            const int n0      = t * 16;
            const int wordIdx = n0 >> 5;
            const int hi      = (n0 & 16) ? 16 : 0;
            v8f acc = {0.f, 0.f, 0.f, 0.f, 0.f, 0.f, 0.f, 0.f};
            for (int oc = 0; oc < NTOK; oc += 32) {
                // A operand: two contiguous 8-half runs (K=kb..kb+7, kb+16..kb+23)
                const _Float16* ap = abase + (areal ? oc : 0);
                V16U A;
                A.h8[0] = *(const v8h*)(ap + kb);
                A.h8[1] = *(const v8h*)(ap + kb + 16);
                // B operand: one membership word per lane (row oc+lane), then
                // 16 shuffles transpose; bits -> f16 {0,1} via *0x3C00 packing.
                int  r    = sIdx[oc + lane];
                u32  wd   = memberW[(bo + r) * NWRD + wordIdx];
                u32  bits = (wd >> hi) & 0xFFFFu;
                V16U Bm;
#pragma unroll
                for (int hp = 0; hp < 8; ++hp) {
                    int h0 = 2 * hp, h1 = 2 * hp + 1;
                    int k0 = kb + ((h0 < 8) ? h0 : h0 + 8);
                    int k1 = kb + ((h1 < 8) ? h1 : h1 + 8);
                    u32 s0 = (u32)__shfl((int)bits, k0, 32);
                    u32 s1 = (u32)__shfl((int)bits, k1, 32);
                    u32 b0 = ((s0 >> col) & 1u) * 0x3C00u;   // f16 1.0
                    u32 b1 = ((s1 >> col) & 1u) * 0x3C00u;
                    Bm.w[hp] = b0 | (b1 << 16);
                }
                acc = __builtin_amdgcn_wmma_f32_16x16x32_f16(
                    false, A.v, false, Bm.v, (short)0, acc, false, false);
            }
            if (lane < 16) {
                idp1W[bo + n0 + lane] = acc[0];   // M=0 row
                ttlW [bo + n0 + lane] = acc[1];   // M=1 row
            }
        }
    }
    __threadfence_block();
    __syncthreads();

    // ------- Phase F: finalize token data + per-batch chain id -------
    float* OD = outData + (size_t)b * NTOK * 9;
    for (int n = tid; n < NTOK; n += THREADS) {
        const float* tk = T + n*9;
        float s = tk[0], fs = tk[3], fe = tk[4], As = tk[5], Ae = tk[6],
              ps = tk[7], pe = tk[8];
        int  idp1 = (int)rintf(idp1W[bo + n]);
        bool inq  = (idp1 > 0);
        cidW[bo + n] = idp1 - 1;
        float new_s = inq ? sqrtf(ttlW[bo + n]) : s;
        int  pd   = predW[bo + n];
        bool mask = inq && (pd >= 0);
        int  p    = mask ? pd : 0;
        float fep = T[p*9+4], aep = T[p*9+6], pep = T[p*9+8];
        float favg = 0.5f * (fep + fs);
        float aavg = 0.5f * (aep + As);
        float corr = wrapf(ps - pep);
        OD[n*9+0] = new_s;
        OD[n*9+1] = tk[1];
        OD[n*9+2] = tk[2];
        OD[n*9+3] = mask ? favg : fs;
        OD[n*9+4] = fe;                       // default; scatter below
        OD[n*9+5] = mask ? aavg : As;
        OD[n*9+6] = Ae;
        OD[n*9+7] = mask ? (ps - 0.5f * corr) : ps;
        OD[n*9+8] = pe;
    }
    __threadfence_block();
    __syncthreads();
    // Conflict-free scatters (qualifying chains are disjoint, preds unique).
    for (int n = tid; n < NTOK; n += THREADS) {
        int idp1 = (int)rintf(idp1W[bo + n]);
        int pd   = predW[bo + n];
        if (idp1 > 0 && pd >= 0) {
            const float* tk = T + n*9;
            float fs = tk[3], As = tk[5], ps = tk[7];
            float fep = T[pd*9+4], aep = T[pd*9+6], pep = T[pd*9+8];
            OD[pd*9+4] = 0.5f * (fep + fs);
            OD[pd*9+6] = 0.5f * (aep + As);
            OD[pd*9+8] = pep + 0.5f * wrapf(ps - pep);
        }
    }
}

__global__ void chirp_chain_offsets(const int* __restrict__ cidW,
                                    const int* __restrict__ nqW,
                                    int* __restrict__ chainOut)
{
    int i = blockIdx.x * blockDim.x + threadIdx.x;
    if (i >= BB * NTOK) return;
    int b = i >> 12;
    int off = 0;
    for (int x = 0; x < b; ++x) off += nqW[x];
    int c = cidW[i];
    chainOut[i] = (c >= 0) ? (c + off) : -1;
}

extern "C" void kernel_launch(void* const* d_in, const int* in_sizes, int n_in,
                              void* d_out, int out_size, void* d_ws, size_t ws_size,
                              hipStream_t stream) {
    const float* tok = (const float*)d_in[0];          // (B,N,9) f32
    float* outData = (float*)d_out;                    // (B,N,9) f32
    int*   chainOut = (int*)((float*)d_out + (size_t)BB * NTOK * 9); // (B,N) i32

    // Workspace carve (≈ 8.9 MB total)
    char* ws = (char*)d_ws;
    u32*   memberW = (u32*)ws;                 ws += (size_t)BB * NTOK * NWRD * sizeof(u32);
    float* bestW   = (float*)ws;               ws += (size_t)BB * NTOK * sizeof(float);
    int*   predW   = (int*)ws;                 ws += (size_t)BB * NTOK * sizeof(int);
    float* totalW  = (float*)ws;               ws += (size_t)BB * NTOK * sizeof(float);
    int*   lenW    = (int*)ws;                 ws += (size_t)BB * NTOK * sizeof(int);
    float* idp1W   = (float*)ws;               ws += (size_t)BB * NTOK * sizeof(float);
    float* ttlW    = (float*)ws;               ws += (size_t)BB * NTOK * sizeof(float);
    int*   cidW    = (int*)ws;                 ws += (size_t)BB * NTOK * sizeof(int);
    int*   nqW     = (int*)ws;

    chirp_link_batch<<<BB, THREADS, 0, stream>>>(
        tok, outData, memberW, bestW, predW, totalW, lenW, idp1W, ttlW, cidW, nqW);
    chirp_chain_offsets<<<(BB * NTOK + 255) / 256, 256, 0, stream>>>(
        cidW, nqW, chainOut);
}
